// GraphTransformerVAE_53498112639116
// MI455X (gfx1250) — compile-verified, matrix operand order checked by reading, not run on hardware
//
#include <hip/hip_runtime.h>

typedef __attribute__((ext_vector_type(16))) _Float16 v16h;
typedef __attribute__((ext_vector_type(8)))  _Float16 v8h;
typedef __attribute__((ext_vector_type(8)))  float    v8f;

#define CDIV(a, b) (((a) + (b) - 1) / (b))

// ---------------------------------------------------------------------------
// float <-> ordered-uint mapping for atomicMax on floats (radix-sort trick)
// ---------------------------------------------------------------------------
__device__ inline unsigned fflip(float f) {
    unsigned u = __float_as_uint(f);
    unsigned mask = (unsigned)(-(int)(u >> 31)) | 0x80000000u;
    return u ^ mask;
}
__device__ inline float funflip(unsigned u) {
    unsigned mask = ((u >> 31) - 1u) | 0x80000000u;
    return __uint_as_float(u ^ mask);
}

// ---------------------------------------------------------------------------
// Pack f32 weight [srcK x OC] into f16 WMMA B-fragment order, zero-padding
// K rows up to KP. Layout: [kc][ct][lane][j] with 16 contiguous f16 per lane,
// so a B fragment is a single contiguous 32-byte read.
//   k(j,half) = kc*32 + ((j>>3)<<4 | half<<3 | (j&7)),  n = ct*16 + (lane&15)
// ---------------------------------------------------------------------------
__global__ void pack_w(const float* __restrict__ src, _Float16* __restrict__ dst,
                       int srcK, int KP, int OC) {
    int idx = blockIdx.x * blockDim.x + threadIdx.x;
    if (idx >= KP * OC) return;
    int j    = idx & 15;
    int lane = (idx >> 4) & 31;
    int tile = idx >> 9;
    int nct  = OC >> 4;
    int ct   = tile % nct;
    int kc   = tile / nct;
    int half = lane >> 4;
    int n    = ct * 16 + (lane & 15);
    int k    = kc * 32 + (((j >> 3) << 4) | (half << 3) | (j & 7));
    dst[idx] = (k < srcK) ? (_Float16)src[(size_t)k * OC + n] : (_Float16)0.f;
}

// plain f32 -> f16 cast
__global__ void cast_f16(const float* __restrict__ src, _Float16* __restrict__ dst, int n) {
    int idx = blockIdx.x * blockDim.x + threadIdx.x;
    if (idx < n) dst[idx] = (_Float16)src[idx];
}

__global__ void zero_f32(float* __restrict__ p, int n) {
    int idx = blockIdx.x * blockDim.x + threadIdx.x;
    if (idx < n) p[idx] = 0.f;
}

// ---------------------------------------------------------------------------
// WMMA GEMM:  C[N x OC] = A_f16[N x KP] @ Wpacked + bias
// One wave per 16-row tile. KP in {32,64}, OC in {16,32,64} (compile-time).
// A fragment = two contiguous b128 loads per k-chunk; B fragment = one
// contiguous 32B load from the packed weight buffer.
// ---------------------------------------------------------------------------
template <int KP, int OC>
__global__ __launch_bounds__(128)
void wmma_gemm(const _Float16* __restrict__ A, const _Float16* __restrict__ Wp,
               const float* __restrict__ bias, float* __restrict__ Cout, int N) {
    constexpr int NKC = KP / 32;
    constexpr int NCT = OC / 16;
    int wave = blockIdx.x * (blockDim.x >> 5) + (threadIdx.x >> 5);
    int lane = threadIdx.x & 31;
    int row0 = wave * 16;
    if (row0 >= N) return;                    // wave-uniform: EXEC stays all-ones
    int half = lane >> 4;
    int mrow = lane & 15;

    v16h afrag[NKC];
#pragma unroll
    for (int kc = 0; kc < NKC; ++kc) {
        const _Float16* ap = A + (size_t)(row0 + mrow) * KP + kc * 32 + half * 8;
        v8h lo = *(const v8h*)(ap);           // k = half*8 + 0..7
        v8h hi = *(const v8h*)(ap + 16);      // k = 16 + half*8 + 0..7
        afrag[kc] = __builtin_shufflevector(lo, hi, 0, 1, 2, 3, 4, 5, 6, 7,
                                            8, 9, 10, 11, 12, 13, 14, 15);
    }

#pragma unroll
    for (int ct = 0; ct < NCT; ++ct) {
        v8f c = {};
#pragma unroll
        for (int kc = 0; kc < NKC; ++kc) {
            v16h b = *(const v16h*)(Wp + (((size_t)kc * NCT + ct) * 32 + lane) * 16);
            c = __builtin_amdgcn_wmma_f32_16x16x32_f16(
                    false, afrag[kc], false, b, (short)0, c, false, false);
        }
        float bb = bias[ct * 16 + mrow];
#pragma unroll
        for (int i = 0; i < 8; ++i) {
            int r = row0 + half * 8 + i;
            Cout[(size_t)r * OC + ct * 16 + mrow] = c[i] + bb;
        }
    }
}

// ---------------------------------------------------------------------------
// Edge pass 1: logits[e,h] = (q[dst]·k[src])/4 per head; atomicMax into m
// ---------------------------------------------------------------------------
__global__ void edge_logits(const int* __restrict__ src, const int* __restrict__ dst,
                            const float* __restrict__ q, const float* __restrict__ k,
                            float* __restrict__ logits, unsigned* __restrict__ mI, int E4) {
    int idx = blockIdx.x * blockDim.x + threadIdx.x;
    if (idx >= E4) return;
    int e = idx >> 2, h = idx & 3;
    int s = src[e], d = dst[e];
    const float4* qp = (const float4*)(q + (size_t)d * 64 + h * 16);
    const float4* kp = (const float4*)(k + (size_t)s * 64 + h * 16);
    float acc = 0.f;
#pragma unroll
    for (int i = 0; i < 4; ++i) {
        float4 a = qp[i], b = kp[i];
        acc += a.x * b.x + a.y * b.y + a.z * b.z + a.w * b.w;
    }
    acc *= 0.25f;                             // 1/sqrt(16)
    logits[idx] = acc;
    atomicMax(mI + (size_t)d * 4 + h, fflip(acc));
}

// ---------------------------------------------------------------------------
// Edge pass 2a: p = exp(logit - m[dst]); overwrite logits with p; s[dst] += p
// ---------------------------------------------------------------------------
__global__ void edge_softmax(const int* __restrict__ dst, float* __restrict__ logits,
                             const unsigned* __restrict__ mI, float* __restrict__ ssum, int E4) {
    int idx = blockIdx.x * blockDim.x + threadIdx.x;
    if (idx >= E4) return;
    int e = idx >> 2, h = idx & 3;
    int d = dst[e];
    float p = __expf(logits[idx] - funflip(mI[(size_t)d * 4 + h]));
    logits[idx] = p;
    atomicAdd(ssum + (size_t)d * 4 + h, p);
}

// ---------------------------------------------------------------------------
// Edge pass 2b: acc[dst][c] += p[e,h(c)] * v[src][c], one thread per (e, c)
// (coalesced v reads and coalesced atomics across c)
// ---------------------------------------------------------------------------
__global__ void edge_scatter(const int* __restrict__ src, const int* __restrict__ dst,
                             const float* __restrict__ pbuf, const float* __restrict__ v,
                             float* __restrict__ acc, int E64) {
    int idx = blockIdx.x * blockDim.x + threadIdx.x;
    if (idx >= E64) return;
    int e = idx >> 6, c = idx & 63, h = c >> 4;
    int s = src[e], d = dst[e];
    float p = pbuf[(size_t)e * 4 + h];
    atomicAdd(acc + (size_t)d * 64 + c, p * v[(size_t)s * 64 + c]);
}

// ---------------------------------------------------------------------------
// combine: h = acc / (s + 1e-16) + skip
// ---------------------------------------------------------------------------
__global__ void combine(const float* __restrict__ acc, const float* __restrict__ ssum,
                        const float* __restrict__ skip, float* __restrict__ hout, int n64) {
    int idx = blockIdx.x * blockDim.x + threadIdx.x;
    if (idx >= n64) return;
    int n = idx >> 6, c = idx & 63, h = c >> 4;
    hout[idx] = acc[idx] / (ssum[(size_t)n * 4 + h] + 1e-16f) + skip[idx];
}

// ---------------------------------------------------------------------------
// LayerNorm(+optional relu) -> f16 output (padded to CP cols with zeros)
// ---------------------------------------------------------------------------
__global__ void layernorm_f16(const float* __restrict__ in, const float* __restrict__ g,
                              const float* __restrict__ b, _Float16* __restrict__ outh,
                              int N, int C, int CP, int relu) {
    int n = blockIdx.x * blockDim.x + threadIdx.x;
    if (n >= N) return;
    const float4* p4 = (const float4*)(in + (size_t)n * C);
    int C4 = C >> 2;
    float mu = 0.f;
    for (int c = 0; c < C4; ++c) {
        float4 t = p4[c];
        mu += t.x + t.y + t.z + t.w;
    }
    mu /= (float)C;
    float var = 0.f;
    for (int c = 0; c < C4; ++c) {
        float4 t = p4[c];
        float a = t.x - mu, bb = t.y - mu, cc = t.z - mu, dd = t.w - mu;
        var += a * a + bb * bb + cc * cc + dd * dd;
    }
    var /= (float)C;
    float rs = rsqrtf(var + 1e-5f);
    const float* p = in + (size_t)n * C;
    _Float16* o = outh + (size_t)n * CP;
    for (int c = 0; c < CP; ++c) {
        float y = 0.f;
        if (c < C) {
            y = (p[c] - mu) * rs * g[c] + b[c];
            if (relu) y = fmaxf(y, 0.f);
        }
        o[c] = (_Float16)y;
    }
}

// ---------------------------------------------------------------------------
// reparameterize: z = mu + eps * exp(0.5*logvar) -> f16
// ---------------------------------------------------------------------------
__global__ void reparam(const float* __restrict__ mu, const float* __restrict__ lv,
                        const float* __restrict__ eps, _Float16* __restrict__ zh, int n) {
    int idx = blockIdx.x * blockDim.x + threadIdx.x;
    if (idx >= n) return;
    zh[idx] = (_Float16)(mu[idx] + eps[idx] * __expf(0.5f * lv[idx]));
}

// ---------------------------------------------------------------------------
extern "C" void kernel_launch(void* const* d_in, const int* in_sizes, int n_in,
                              void* d_out, int out_size, void* d_ws, size_t ws_size,
                              hipStream_t stream) {
    const int N   = in_sizes[0] / 64;   // 50000
    const int E   = in_sizes[1] / 2;    // 800000
    const int E4  = E * 4;
    const int E64 = E * 64;
    const int N64 = N * 64;

    const float* x    = (const float*)d_in[0];
    const int*   ei   = (const int*)d_in[1];
    const int*   esrc = ei;
    const int*   edst = ei + E;
    const float* eps  = (const float*)d_in[2];

    const float* c1Wq = (const float*)d_in[3];  const float* c1bq = (const float*)d_in[4];
    const float* c1Wk = (const float*)d_in[5];  const float* c1bk = (const float*)d_in[6];
    const float* c1Wv = (const float*)d_in[7];  const float* c1bv = (const float*)d_in[8];
    const float* c1Ws = (const float*)d_in[9];  const float* c1bs = (const float*)d_in[10];
    const float* ln1g = (const float*)d_in[11]; const float* ln1b = (const float*)d_in[12];
    const float* c2Wq = (const float*)d_in[13]; const float* c2bq = (const float*)d_in[14];
    const float* c2Wk = (const float*)d_in[15]; const float* c2bk = (const float*)d_in[16];
    const float* c2Wv = (const float*)d_in[17]; const float* c2bv = (const float*)d_in[18];
    const float* c2Ws = (const float*)d_in[19]; const float* c2bs = (const float*)d_in[20];
    const float* ln2g = (const float*)d_in[21]; const float* ln2b = (const float*)d_in[22];
    const float* muW  = (const float*)d_in[23]; const float* mub  = (const float*)d_in[24];
    const float* lvW  = (const float*)d_in[25]; const float* lvb  = (const float*)d_in[26];
    const float* fc1W = (const float*)d_in[27]; const float* fc1b = (const float*)d_in[28];
    const float* dn1g = (const float*)d_in[29]; const float* dn1b = (const float*)d_in[30];
    const float* fc2W = (const float*)d_in[31]; const float* fc2b = (const float*)d_in[32];
    const float* dn2g = (const float*)d_in[33]; const float* dn2b = (const float*)d_in[34];
    const float* fc3W = (const float*)d_in[35]; const float* fc3b = (const float*)d_in[36];

    // ------------------------- workspace layout -------------------------
    char* ws = (char*)d_ws;
    size_t off = 0;
    auto bump = [&](size_t bytes) { void* p = ws + off; off = (off + bytes + 255) & ~(size_t)255; return p; };

    _Float16* wb    = (_Float16*)bump(45568 * sizeof(_Float16)); // packed f16 weights
    _Float16* xh    = (_Float16*)bump((size_t)N64 * 2);
    float*    q     = (float*)bump((size_t)N64 * 4);
    float*    k     = (float*)bump((size_t)N64 * 4);
    float*    v     = (float*)bump((size_t)N64 * 4);
    float*    skip  = (float*)bump((size_t)N64 * 4);
    float*    acc   = (float*)bump((size_t)N64 * 4);             // acc|mI|ssum contiguous
    unsigned* mI    = (unsigned*)bump((size_t)N * 4 * 4);
    float*    ssum  = (float*)bump((size_t)N * 4 * 4);
    float*    lgts  = (float*)bump((size_t)E4 * 4);
    float*    h     = (float*)bump((size_t)N64 * 4);
    _Float16* hh1   = (_Float16*)bump((size_t)N64 * 2);
    _Float16* hh2   = (_Float16*)bump((size_t)N64 * 2);
    _Float16* zh    = (_Float16*)bump((size_t)N64 * 2);
    float*    d1    = (float*)bump((size_t)N * 32 * 4);
    _Float16* d1h   = (_Float16*)bump((size_t)N * 32 * 2);
    float*    d2    = (float*)bump((size_t)N * 16 * 4);
    _Float16* d2h   = (_Float16*)bump((size_t)N * 32 * 2);       // padded K=32

    _Float16* w_c1q = wb;          _Float16* w_c1k = wb + 4096;
    _Float16* w_c1v = wb + 8192;   _Float16* w_c1s = wb + 12288;
    _Float16* w_c2q = wb + 16384;  _Float16* w_c2k = wb + 20480;
    _Float16* w_c2v = wb + 24576;  _Float16* w_c2s = wb + 28672;
    _Float16* w_mu  = wb + 32768;  _Float16* w_lv  = wb + 36864;
    _Float16* w_f1  = wb + 40960;  // KP=64, OC=32
    _Float16* w_f2  = wb + 43008;  // KP=32, OC=16
    _Float16* w_f3  = wb + 43520;  // KP=32, OC=64 (K rows 16..31 zero)

    float* xhat = (float*)d_out;
    float* muO  = (float*)d_out + (size_t)N64;
    float* lvO  = (float*)d_out + (size_t)2 * N64;

    const int TB = 256;
    dim3 gridE(CDIV(E4, TB));
    dim3 gridE64(CDIV(E64, TB));
    dim3 gridN64(CDIV(N64, TB));
    dim3 gridNode(CDIV(N, TB));
    dim3 gridGemm(CDIV(N / 16, 4));            // 4 waves/block, 1 tile/wave

    // ------------------- weight packing (every call) -------------------
    auto pk = [&](const float* s, _Float16* d, int srcK, int KP, int OC) {
        pack_w<<<dim3(CDIV(KP * OC, TB)), TB, 0, stream>>>(s, d, srcK, KP, OC);
    };
    pk(c1Wq, w_c1q, 64, 64, 64); pk(c1Wk, w_c1k, 64, 64, 64);
    pk(c1Wv, w_c1v, 64, 64, 64); pk(c1Ws, w_c1s, 64, 64, 64);
    pk(c2Wq, w_c2q, 64, 64, 64); pk(c2Wk, w_c2k, 64, 64, 64);
    pk(c2Wv, w_c2v, 64, 64, 64); pk(c2Ws, w_c2s, 64, 64, 64);
    pk(muW,  w_mu,  64, 64, 64); pk(lvW,  w_lv,  64, 64, 64);
    pk(fc1W, w_f1, 64, 64, 32);
    pk(fc2W, w_f2, 32, 32, 16);
    pk(fc3W, w_f3, 16, 32, 64);                 // zero-pad K to 32
    cast_f16<<<gridN64, TB, 0, stream>>>(x, xh, N64);

    // --------------------------- conv layer 1 ---------------------------
    wmma_gemm<64, 64><<<gridGemm, 128, 0, stream>>>(xh, w_c1q, c1bq, q,    N);
    wmma_gemm<64, 64><<<gridGemm, 128, 0, stream>>>(xh, w_c1k, c1bk, k,    N);
    wmma_gemm<64, 64><<<gridGemm, 128, 0, stream>>>(xh, w_c1v, c1bv, v,    N);
    wmma_gemm<64, 64><<<gridGemm, 128, 0, stream>>>(xh, w_c1s, c1bs, skip, N);
    zero_f32<<<dim3(CDIV(N * 72, TB)), TB, 0, stream>>>(acc, N * 72);   // acc+mI+ssum
    edge_logits<<<gridE, TB, 0, stream>>>(esrc, edst, q, k, lgts, mI, E4);
    edge_softmax<<<gridE, TB, 0, stream>>>(edst, lgts, mI, ssum, E4);
    edge_scatter<<<gridE64, TB, 0, stream>>>(esrc, edst, lgts, v, acc, E64);
    combine<<<gridN64, TB, 0, stream>>>(acc, ssum, skip, h, N64);
    layernorm_f16<<<gridNode, TB, 0, stream>>>(h, ln1g, ln1b, hh1, N, 64, 64, 1);

    // --------------------------- conv layer 2 ---------------------------
    wmma_gemm<64, 64><<<gridGemm, 128, 0, stream>>>(hh1, w_c2q, c2bq, q,    N);
    wmma_gemm<64, 64><<<gridGemm, 128, 0, stream>>>(hh1, w_c2k, c2bk, k,    N);
    wmma_gemm<64, 64><<<gridGemm, 128, 0, stream>>>(hh1, w_c2v, c2bv, v,    N);
    wmma_gemm<64, 64><<<gridGemm, 128, 0, stream>>>(hh1, w_c2s, c2bs, skip, N);
    zero_f32<<<dim3(CDIV(N * 72, TB)), TB, 0, stream>>>(acc, N * 72);
    edge_logits<<<gridE, TB, 0, stream>>>(esrc, edst, q, k, lgts, mI, E4);
    edge_softmax<<<gridE, TB, 0, stream>>>(edst, lgts, mI, ssum, E4);
    edge_scatter<<<gridE64, TB, 0, stream>>>(esrc, edst, lgts, v, acc, E64);
    combine<<<gridN64, TB, 0, stream>>>(acc, ssum, skip, h, N64);
    layernorm_f16<<<gridNode, TB, 0, stream>>>(h, ln2g, ln2b, hh2, N, 64, 64, 0);

    // -------------------------- VAE head --------------------------------
    wmma_gemm<64, 64><<<gridGemm, 128, 0, stream>>>(hh2, w_mu, mub, muO, N);
    wmma_gemm<64, 64><<<gridGemm, 128, 0, stream>>>(hh2, w_lv, lvb, lvO, N);
    reparam<<<gridN64, TB, 0, stream>>>(muO, lvO, eps, zh, N64);

    // -------------------------- decoder ---------------------------------
    wmma_gemm<64, 32><<<gridGemm, 128, 0, stream>>>(zh,  w_f1, fc1b, d1, N);
    layernorm_f16<<<gridNode, TB, 0, stream>>>(d1, dn1g, dn1b, d1h, N, 32, 32, 1);
    wmma_gemm<32, 16><<<gridGemm, 128, 0, stream>>>(d1h, w_f2, fc2b, d2, N);
    layernorm_f16<<<gridNode, TB, 0, stream>>>(d2, dn2g, dn2b, d2h, N, 16, 32, 1); // pad K
    wmma_gemm<32, 64><<<gridGemm, 128, 0, stream>>>(d2h, w_f3, fc3b, xhat, N);

    (void)n_in; (void)out_size; (void)ws_size; (void)off;
}
